// CrossTemporalAttention2_16673063043464
// MI455X (gfx1250) — matrix operand order repaired and checked by reading, not compiled, not run on hardware
//
#include <hip/hip_runtime.h>
#include <hip/hip_bf16.h>

typedef __attribute__((ext_vector_type(16))) _Float16 v16h;
typedef __attribute__((ext_vector_type(8)))  float    v8f;
typedef __attribute__((ext_vector_type(4)))  unsigned int v4u;
typedef int v4i_ __attribute__((vector_size(16)));   // matches builtin param type

#define GLOBAL_AS __attribute__((address_space(1)))
#define LDS_AS    __attribute__((address_space(3)))

typedef _Float16 f16;

#define WMMA_F16(a,b,c) __builtin_amdgcn_wmma_f32_16x16x32_f16(false,(a),false,(b),(short)0,(c),false,false)

#define BATCH 2
#define NTOK  4096
#define CDIM  256
#define NH    8
#define DH    32
#define MTOK  1024
#define KSR   1024

#if __has_builtin(__builtin_amdgcn_global_load_async_to_lds_b128)
#define ASYNC_LDS 1
#endif

// 16-byte global -> LDS copy: async DMA on CDNA5 (tracked by ASYNCcnt),
// synchronous fallback otherwise. Both paths assume 16B alignment.
__device__ __forceinline__ void cp16(const f16* g, f16* l) {
#ifdef ASYNC_LDS
  __builtin_amdgcn_global_load_async_to_lds_b128(
      (GLOBAL_AS v4i_*)g,
      (LDS_AS v4i_*)l, 0, 0);
#else
  *(v4u*)l = *(const v4u*)g;
#endif
}

__device__ __forceinline__ void wait_async_lds() {
#ifdef ASYNC_LDS
#if __has_builtin(__builtin_amdgcn_s_wait_asynccnt)
  __builtin_amdgcn_s_wait_asynccnt(0);
#else
  asm volatile("s_wait_asynccnt 0x0" ::: "memory");
#endif
#endif
}

// ---------------------------------------------------------------------------
// Weight prep: f32->f16 conversions; sr_w permuted [co][ci][kh][kw] ->
// [co][p*256+ci] with p = kh*2+kw so the SR conv is a K=1024 GEMM whose
// K-chunks are contiguous in ci for both operands.
// ---------------------------------------------------------------------------
__global__ void k_prep(const float* __restrict__ qw, const float* __restrict__ kvw,
                       const float* __restrict__ projw, const float* __restrict__ srw,
                       f16* __restrict__ wq, f16* __restrict__ wkv,
                       f16* __restrict__ wproj, f16* __restrict__ wsr) {
  int t = blockIdx.x * 256 + threadIdx.x;
  if (t < 65536) {
    wq[t] = (f16)qw[t];
  } else if (t < 65536 + 131072) {
    int i = t - 65536; wkv[i] = (f16)kvw[i];
  } else if (t < 65536 + 131072 + 65536) {
    int i = t - 196608; wproj[i] = (f16)projw[i];
  } else if (t < 524288) {
    int i = t - 262144;
    int co = i >> 10, rem = i & 1023, p = rem >> 8, ci = rem & 255;
    wsr[i] = (f16)srw[((co * 256 + ci) << 2) + p];
  }
}

// ---------------------------------------------------------------------------
// Q projection: y[n][co] = sum_k x[n][k] * w[co][k];  x f32, w f16, y f16.
// One wave = 16 rows x 32 cols; 8 waves/WG cover all 256 cols.
// ---------------------------------------------------------------------------
__global__ void k_gemm_q(const float* __restrict__ x, const f16* __restrict__ w,
                         f16* __restrict__ y) {
  int lane = threadIdx.x & 31, wv = threadIdx.x >> 5;
  int row0 = blockIdx.x * 16, col0 = wv * 32;
  int am = lane & 15;
  int aoff = (lane & 16) ? 16 : 0;
  int bn = lane & 15;
  int bko = (lane & 16) ? 16 : 0;
  v8f c0 = {}, c1 = {};
  const float* arow = x + (size_t)(row0 + am) * CDIM;
  for (int k0 = 0; k0 < CDIM; k0 += 32) {
    __builtin_prefetch(arow + k0 + 32, 0, 0);   // global_prefetch_b8, stream x from HBM
    v16h a, b0, b1;
#pragma unroll
    for (int hh = 0; hh < 8; ++hh) {
      a[hh]     = (f16)arow[k0 + aoff + hh];
      a[hh + 8] = (f16)arow[k0 + aoff + 8 + hh];
    }
    const f16* w0 = w + (size_t)(col0 + bn) * CDIM + k0 + bko;
    const f16* w1 = w + (size_t)(col0 + 16 + bn) * CDIM + k0 + bko;
#pragma unroll
    for (int hh = 0; hh < 16; ++hh) { b0[hh] = w0[hh]; b1[hh] = w1[hh]; }
    c0 = WMMA_F16(a, b0, c0);
    c1 = WMMA_F16(a, b1, c1);
  }
  int g = lane & 15, rb = (lane & 16) ? 8 : 0;
#pragma unroll
  for (int r = 0; r < 8; ++r) {
    int row = row0 + rb + r;
    y[(size_t)row * CDIM + col0 + g]      = (f16)c0[r];
    y[(size_t)row * CDIM + col0 + 16 + g] = (f16)c1[r];
  }
}

// ---------------------------------------------------------------------------
// SR reduce (conv k=2 s=2 as K=1024 GEMM) + bias + LayerNorm, output f16.
// WG owns 16 full output rows (256 cols), so LN stats via shuffles + LDS.
// ---------------------------------------------------------------------------
__global__ void k_sr_ln(const float* __restrict__ x, const f16* __restrict__ wsr,
                        const float* __restrict__ srb, const float* __restrict__ lng,
                        const float* __restrict__ lnb, f16* __restrict__ xr) {
  __shared__ float s_sum[16], s_sq[16];
  int lane = threadIdx.x & 31, wv = threadIdx.x >> 5;
  if (threadIdx.x < 16) { s_sum[threadIdx.x] = 0.f; s_sq[threadIdx.x] = 0.f; }
  __syncthreads();
  int row0 = blockIdx.x * 16, col0 = wv * 32;
  int am = lane & 15;
  int aoff = (lane & 16) ? 16 : 0;
  int bn = lane & 15;
  int bko = (lane & 16) ? 16 : 0;
  int gr = row0 + am;
  int bb = gr >> 10, m = gr & 1023;
  int oh = m >> 5, ow = m & 31;
  v8f c0 = {}, c1 = {};
  for (int k0 = 0; k0 < KSR; k0 += 32) {
    int p = k0 >> 8, kh = p >> 1, kw = p & 1, ci0 = k0 & 255;
    int pix = ((oh * 2 + kh) << 6) + ow * 2 + kw;
    const float* arow = x + ((size_t)(bb * NTOK + pix)) * CDIM + ci0;
    __builtin_prefetch(arow + 32, 0, 0);
    v16h a, b0, b1;
#pragma unroll
    for (int hh = 0; hh < 8; ++hh) {
      a[hh]     = (f16)arow[aoff + hh];
      a[hh + 8] = (f16)arow[aoff + 8 + hh];
    }
    const f16* w0 = wsr + (size_t)(col0 + bn) * KSR + k0 + bko;
    const f16* w1 = wsr + (size_t)(col0 + 16 + bn) * KSR + k0 + bko;
#pragma unroll
    for (int hh = 0; hh < 16; ++hh) { b0[hh] = w0[hh]; b1[hh] = w1[hh]; }
    c0 = WMMA_F16(a, b0, c0);
    c1 = WMMA_F16(a, b1, c1);
  }
  int g = lane & 15, rb = (lane & 16) ? 8 : 0;
  float bias0 = srb[col0 + g], bias1 = srb[col0 + 16 + g];
#pragma unroll
  for (int r = 0; r < 8; ++r) { c0[r] += bias0; c1[r] += bias1; }
  // per-row partial sums over this wave's 32 columns
#pragma unroll
  for (int r = 0; r < 8; ++r) {
    float s = c0[r] + c1[r];
    float q = c0[r] * c0[r] + c1[r] * c1[r];
    for (int d = 1; d < 16; d <<= 1) {
      s += __shfl_xor(s, d, 32);
      q += __shfl_xor(q, d, 32);
    }
    if (g == r) { atomicAdd(&s_sum[rb + r], s); atomicAdd(&s_sq[rb + r], q); }
  }
  __syncthreads();
  float gg0 = lng[col0 + g], gg1 = lng[col0 + 16 + g];
  float be0 = lnb[col0 + g], be1 = lnb[col0 + 16 + g];
#pragma unroll
  for (int r = 0; r < 8; ++r) {
    int rowl = rb + r;
    float mu = s_sum[rowl] * (1.f / 256.f);
    float var = s_sq[rowl] * (1.f / 256.f) - mu * mu;
    float inv = rsqrtf(var + 1e-5f);
    int row = row0 + rowl;
    xr[(size_t)row * CDIM + col0 + g]      = (f16)((c0[r] - mu) * inv * gg0 + be0);
    xr[(size_t)row * CDIM + col0 + 16 + g] = (f16)((c1[r] - mu) * inv * gg1 + be1);
  }
}

// ---------------------------------------------------------------------------
// KV projection. transposed=0: dst[b][h][m][dh] (keys); transposed=1:
// dst[b][h][dh][m] (values, pre-transposed for the PV A-operand).
// ---------------------------------------------------------------------------
__global__ void k_gemm_kv(const f16* __restrict__ xr, const f16* __restrict__ wkv,
                          int wrow0, f16* __restrict__ dst, int transposed) {
  int lane = threadIdx.x & 31, wv = threadIdx.x >> 5;
  int row0 = blockIdx.x * 16, col0 = wv * 32;
  int am = lane & 15;
  int aoff = (lane & 16) ? 16 : 0;
  int bn = lane & 15;
  int bko = (lane & 16) ? 16 : 0;
  v8f c0 = {}, c1 = {};
  const f16* arow = xr + (size_t)(row0 + am) * CDIM;
  for (int k0 = 0; k0 < CDIM; k0 += 32) {
    v16h a, b0, b1;
#pragma unroll
    for (int hh = 0; hh < 8; ++hh) {
      a[hh]     = arow[k0 + aoff + hh];
      a[hh + 8] = arow[k0 + aoff + 8 + hh];
    }
    const f16* w0 = wkv + (size_t)(wrow0 + col0 + bn) * CDIM + k0 + bko;
    const f16* w1 = wkv + (size_t)(wrow0 + col0 + 16 + bn) * CDIM + k0 + bko;
#pragma unroll
    for (int hh = 0; hh < 16; ++hh) { b0[hh] = w0[hh]; b1[hh] = w1[hh]; }
    c0 = WMMA_F16(a, b0, c0);
    c1 = WMMA_F16(a, b1, c1);
  }
  int g = lane & 15, rb = (lane & 16) ? 8 : 0;
  int cA = col0 + g, cB = col0 + 16 + g;
  int hA = cA >> 5, dA = cA & 31, hB = cB >> 5, dB = cB & 31;
#pragma unroll
  for (int r = 0; r < 8; ++r) {
    int row = row0 + rb + r;
    int bidx = row >> 10, mm = row & 1023;
    if (!transposed) {
      dst[(((size_t)bidx * NH + hA) * MTOK + mm) * DH + dA] = (f16)c0[r];
      dst[(((size_t)bidx * NH + hB) * MTOK + mm) * DH + dB] = (f16)c1[r];
    } else {
      dst[(((size_t)bidx * NH + hA) * DH + dA) * MTOK + mm] = (f16)c0[r];
      dst[(((size_t)bidx * NH + hB) * DH + dB) * MTOK + mm] = (f16)c1[r];
    }
  }
}

// ---------------------------------------------------------------------------
// Flash cross-attention, S^T trick: S^T = K x Q^T so the score C-layout maps
// to the P^T B-operand with one shfl_xor(16) exchange; O^T = V^T x P^T.
// One wave = 16 query rows over all M=1024 keys; 4 waves per WG share K/V
// tiles staged in LDS by async global->LDS DMA, double-buffered.
// ---------------------------------------------------------------------------
__global__ void k_attn(const f16* __restrict__ q, const f16* __restrict__ kb,
                       const f16* __restrict__ vt, f16* __restrict__ ab) {
  __shared__ __align__(16) f16 kt[2][32 * DH];   // [buf][key][dh]
  __shared__ __align__(16) f16 vs[2][DH * 32];   // [buf][dh][key]
  int tid = threadIdx.x;
  int lane = tid & 31, wv = tid >> 5;
  int bh = blockIdx.x >> 6;        // 0..15 = (b,h)
  int tile = blockIdx.x & 63;
  int b = bh >> 3, h = bh & 7;
  int row0 = tile * 64 + wv * 16;
  int n16 = lane & 15;
  bool hi = (lane & 16) != 0;
  int aoff = hi ? 16 : 0;

  const f16* kbase = kb + ((size_t)(b * NH + h)) * MTOK * DH;
  const f16* vbase = vt + ((size_t)(b * NH + h)) * DH * MTOK;

  // stage one 32-key chunk (K tile 2KB + V^T tile 2KB) into LDS buffer p
  int sdh = tid >> 2, sseg = tid & 3;
  auto stage = [&](int kc, int p) {
    cp16(kbase + (size_t)kc * DH + tid * 8, &kt[p][tid * 8]);                       // contiguous keys
    cp16(vbase + (size_t)sdh * MTOK + kc + sseg * 8, &vs[p][sdh * 32 + sseg * 8]);  // 32 strided dh rows
  };
  stage(0, 0);

  // Q^T as B-operand: column = qrow, K rows = dh (lanes<16: 0-15, hi: 16-31)
  v16h bq;
  {
    const f16* qp = q + ((size_t)(b * NTOK + row0 + n16)) * CDIM + h * DH + (hi ? 16 : 0);
#pragma unroll
    for (int hh = 0; hh < 16; ++hh) bq[hh] = qp[hh];
  }

  v8f o0 = {}, o1 = {};
  float mr = -1e30f, lr = 0.f;
  const float scale = 0.17677669529663687f;  // 32^-0.5

  int p = 0;
  for (int kc = 0; kc < MTOK; kc += 32) {
    wait_async_lds();      // my async copies for chunk kc complete
    __syncthreads();       // whole WG's copies done; prev buffer free
    if (kc + 32 < MTOK) stage(kc + 32, p ^ 1);

    const f16* ktp = &kt[p][0];
    const f16* vtp = &vs[p][0];

    // K tiles as A-operand: row = key, K = dh (from LDS)
    v16h ak0, ak1;
    {
      const f16* kp0 = ktp + n16 * DH;
      const f16* kp1 = ktp + (16 + n16) * DH;
#pragma unroll
      for (int hh = 0; hh < 8; ++hh) {
        ak0[hh]     = kp0[aoff + hh];
        ak0[hh + 8] = kp0[aoff + 8 + hh];
        ak1[hh]     = kp1[aoff + hh];
        ak1[hh + 8] = kp1[aoff + 8 + hh];
      }
    }
    v8f z = {};
    v8f s0 = WMMA_F16(ak0, bq, z);   // S^T keys kc..kc+15
    v8f s1 = WMMA_F16(ak1, bq, z);   // S^T keys kc+16..kc+31

    float cm = -1e30f;
#pragma unroll
    for (int r = 0; r < 8; ++r) {
      s0[r] *= scale; s1[r] *= scale;
      cm = fmaxf(cm, fmaxf(s0[r], s1[r]));
    }
    cm = fmaxf(cm, __shfl_xor(cm, 16, 32));
    float mn = fmaxf(mr, cm);
    float corr = __expf(mr - mn);
    float csum = 0.f;
#pragma unroll
    for (int r = 0; r < 8; ++r) {
      s0[r] = __expf(s0[r] - mn);
      s1[r] = __expf(s1[r] - mn);
      csum += s0[r] + s1[r];
    }
    csum += __shfl_xor(csum, 16, 32);
    lr = lr * corr + csum;
    mr = mn;
#pragma unroll
    for (int r = 0; r < 8; ++r) { o0[r] *= corr; o1[r] *= corr; }

    // P^T as B-operand: lanes<16 hold keys kc..kc+15, hi lanes kc+16..kc+31
    v16h bp;
#pragma unroll
    for (int r = 0; r < 8; ++r) {
      float x0 = __shfl_xor(s0[r], 16, 32);
      float x1v = __shfl_xor(s1[r], 16, 32);
      bp[r]     = (f16)(hi ? x1v : s0[r]);
      bp[r + 8] = (f16)(hi ? s1[r] : x0);
    }

    // V^T tiles as A-operand: row = dh, K = keys (from LDS)
    v16h av0, av1;
    {
      const f16* vp0 = vtp + n16 * 32;          // dh 0-15
      const f16* vp1 = vtp + (16 + n16) * 32;   // dh 16-31
#pragma unroll
      for (int hh = 0; hh < 8; ++hh) {
        av0[hh]     = vp0[aoff + hh];
        av0[hh + 8] = vp0[aoff + 8 + hh];
        av1[hh]     = vp1[aoff + hh];
        av1[hh + 8] = vp1[aoff + 8 + hh];
      }
    }
    o0 = WMMA_F16(av0, bp, o0);
    o1 = WMMA_F16(av1, bp, o1);
    p ^= 1;
  }

  float inv = 1.f / lr;
  int rb = hi ? 8 : 0;
  f16* op = ab + ((size_t)(b * NTOK + row0 + n16)) * CDIM + h * DH;
#pragma unroll
  for (int r = 0; r < 8; ++r) {
    op[rb + r]      = (f16)(o0[r] * inv);   // dh = rb+r
    op[16 + rb + r] = (f16)(o1[r] * inv);   // dh = 16+rb+r
  }
}

// ---------------------------------------------------------------------------
// Output projection + bias, f32 output.
// ---------------------------------------------------------------------------
__global__ void k_gemm_proj(const f16* __restrict__ a_, const f16* __restrict__ w,
                            const float* __restrict__ bias, float* __restrict__ y) {
  int lane = threadIdx.x & 31, wv = threadIdx.x >> 5;
  int row0 = blockIdx.x * 16, col0 = wv * 32;
  int am = lane & 15;
  int aoff = (lane & 16) ? 16 : 0;
  int bn = lane & 15;
  int bko = (lane & 16) ? 16 : 0;
  v8f c0 = {}, c1 = {};
  const f16* arow = a_ + (size_t)(row0 + am) * CDIM;
  for (int k0 = 0; k0 < CDIM; k0 += 32) {
    v16h a, b0, b1;
#pragma unroll
    for (int hh = 0; hh < 8; ++hh) {
      a[hh]     = arow[k0 + aoff + hh];
      a[hh + 8] = arow[k0 + aoff + 8 + hh];
    }
    const f16* w0 = w + (size_t)(col0 + bn) * CDIM + k0 + bko;
    const f16* w1 = w + (size_t)(col0 + 16 + bn) * CDIM + k0 + bko;
#pragma unroll
    for (int hh = 0; hh < 16; ++hh) { b0[hh] = w0[hh]; b1[hh] = w1[hh]; }
    c0 = WMMA_F16(a, b0, c0);
    c1 = WMMA_F16(a, b1, c1);
  }
  int g = lane & 15, rb = (lane & 16) ? 8 : 0;
  float bias0 = bias[col0 + g], bias1 = bias[col0 + 16 + g];
#pragma unroll
  for (int r = 0; r < 8; ++r) {
    int row = row0 + rb + r;
    y[(size_t)row * CDIM + col0 + g]      = c0[r] + bias0;
    y[(size_t)row * CDIM + col0 + 16 + g] = c1[r] + bias1;
  }
}

extern "C" void kernel_launch(void* const* d_in, const int* in_sizes, int n_in,
                              void* d_out, int out_size, void* d_ws, size_t ws_size,
                              hipStream_t stream) {
  (void)in_sizes; (void)n_in; (void)out_size; (void)ws_size;
  const float* x1     = (const float*)d_in[0];
  const float* x2     = (const float*)d_in[1];
  const float* q_w    = (const float*)d_in[2];
  const float* kv_w   = (const float*)d_in[3];
  const float* sr_w   = (const float*)d_in[4];
  const float* sr_b   = (const float*)d_in[5];
  const float* ln_g   = (const float*)d_in[6];
  const float* ln_b   = (const float*)d_in[7];
  const float* proj_w = (const float*)d_in[8];
  const float* proj_b = (const float*)d_in[9];

  char* ws = (char*)d_ws;
  size_t off = 0;
  f16* wq    = (f16*)(ws + off); off += 256 * 256 * 2;        // 131072
  f16* wkv   = (f16*)(ws + off); off += 512 * 256 * 2;        // 262144
  f16* wproj = (f16*)(ws + off); off += 256 * 256 * 2;        // 131072
  f16* wsr   = (f16*)(ws + off); off += 256 * 1024 * 2;       // 524288
  f16* qbuf  = (f16*)(ws + off); off += (size_t)BATCH * NTOK * CDIM * 2;  // 4 MB
  f16* x1r   = (f16*)(ws + off); off += (size_t)BATCH * MTOK * CDIM * 2;  // 1 MB
  f16* x2r   = (f16*)(ws + off); off += (size_t)BATCH * MTOK * CDIM * 2;  // 1 MB
  f16* kbuf  = (f16*)(ws + off); off += (size_t)BATCH * NH * MTOK * DH * 2;
  f16* vbufT = (f16*)(ws + off); off += (size_t)BATCH * NH * DH * MTOK * 2;
  f16* abuf  = (f16*)(ws + off); off += (size_t)BATCH * NTOK * CDIM * 2;

  k_prep<<<2048, 256, 0, stream>>>(q_w, kv_w, proj_w, sr_w, wq, wkv, wproj, wsr);
  k_gemm_q<<<512, 256, 0, stream>>>(x1, wq, qbuf);
  k_sr_ln<<<128, 256, 0, stream>>>(x1, wsr, sr_b, ln_g, ln_b, x1r);
  k_sr_ln<<<128, 256, 0, stream>>>(x2, wsr, sr_b, ln_g, ln_b, x2r);
  k_gemm_kv<<<128, 256, 0, stream>>>(x2r, wkv, 0,   kbuf,  0);  // keys from x2r
  k_gemm_kv<<<128, 256, 0, stream>>>(x1r, wkv, 256, vbufT, 1);  // values from x1r
  k_attn<<<1024, 128, 0, stream>>>(qbuf, kbuf, vbufT, abuf);
  k_gemm_proj<<<512, 256, 0, stream>>>(abuf, wproj, proj_b, (float*)d_out);
}
